// PerformerAttention_15135464751438
// MI455X (gfx1250) — matrix operand order repaired and checked by reading, not compiled, not run on hardware
//
#include <hip/hip_runtime.h>

// ---------------------------------------------------------------------------
// Performer attention for MI455X (gfx1250, wave32, WMMA bf16 path).
// Big GEMMs use double-buffered LDS staging via async-to-LDS (ASYNCcnt).
// ---------------------------------------------------------------------------

namespace {
constexpr int kH    = 16;
constexpr int kDh   = 64;
constexpr int kM    = 64;           // nb_features == Dh
constexpr int kB    = 4;
constexpr int kC    = 8;
constexpr int kP    = 512;
constexpr int kD    = 1024;
constexpr int kS    = kP * kC;      // 4096
constexpr int kRows = kB * kS;      // 16384
constexpr int kNqkv = 3 * kD;       // 3072
constexpr float kEps = 1e-3f;
constexpr int kLdaS = 40;           // padded LDS row length (halves): 80B rows
}

typedef __attribute__((ext_vector_type(16))) __bf16 v16bf;
typedef __attribute__((ext_vector_type(8)))  float  v8f;

#if __has_builtin(__builtin_amdgcn_global_load_async_to_lds_b128) && \
    __has_builtin(__builtin_amdgcn_s_wait_asynccnt)
#define ASYNC_LDS_OK 1
typedef int vint4 __attribute__((vector_size(16)));        // raw clang vector
typedef __attribute__((address_space(1))) vint4 gbl_vint4; // "__device__" (AS1)
typedef __attribute__((address_space(3))) vint4 lds_vint4; // LDS (AS3)
#endif

union Frag16 {
  v16bf v;
  unsigned short h[16];
  uint4 q[2];
};

__device__ __forceinline__ unsigned short f2bf(float x) {
  union { float f; unsigned int u; } t; t.f = x;
  unsigned int r = t.u + 0x7FFFu + ((t.u >> 16) & 1u);
  return (unsigned short)(r >> 16);
}

__device__ __forceinline__ v8f wmma_bf16(v16bf a, v16bf b, v8f c) {
  return __builtin_amdgcn_wmma_f32_16x16x32_bf16(false, a, false, b,
                                                 (short)0, c, false, false);
}

// ---- direct-from-global fragment loaders (NT layouts) ----------------------
__device__ __forceinline__ v16bf load_a_frag(const unsigned short* __restrict__ A,
                                             int lda, long m0, int k0, int lane) {
  Frag16 f;
  long m = m0 + (lane & 15);
  int  kb = k0 + ((lane & 16) ? 16 : 0);
  const unsigned short* p = A + m * (long)lda + kb;
  f.q[0] = *(const uint4*)(p);
  f.q[1] = *(const uint4*)(p + 8);
  return f.v;
}

__device__ __forceinline__ v16bf load_b_frag_nt(const unsigned short* __restrict__ W,
                                                int ldw, int n0, int k0, int lane) {
  Frag16 f;
  int n  = n0 + (lane & 15);
  int kb = k0 + ((lane & 16) ? 16 : 0);
  const unsigned short* p = W + (long)n * ldw + kb;
  f.q[0] = *(const uint4*)(p);
  f.q[1] = *(const uint4*)(p + 8);
  return f.v;
}

// ---- LDS fragment loaders (tiles stored [row][kLdaS]) ----------------------
__device__ __forceinline__ v16bf lds_a_frag(const unsigned short* base, int r0, int lane) {
  Frag16 f;
  const unsigned short* p = base + (r0 + (lane & 15)) * kLdaS + ((lane & 16) ? 16 : 0);
  f.q[0] = *(const uint4*)(p);
  f.q[1] = *(const uint4*)(p + 8);
  return f.v;
}
__device__ __forceinline__ v16bf lds_b_frag(const unsigned short* base, int r0, int lane) {
  Frag16 f;
  const unsigned short* p = base + (r0 + (lane & 15)) * kLdaS + ((lane & 16) ? 16 : 0);
  f.q[0] = *(const uint4*)(p);
  f.q[1] = *(const uint4*)(p + 8);
  return f.v;
}

// ---- transposed / special loaders for the kvs stage ------------------------
__device__ __forceinline__ v16bf load_a_frag_t(const unsigned short* __restrict__ Kp,
                                               int ld, int j0, int s0, int lane) {
  Frag16 f;
  int j  = j0 + (lane & 15);
  int sb = s0 + ((lane & 16) ? 16 : 0);
#pragma unroll
  for (int i = 0; i < 16; ++i) {
    int kk = sb + i;
    f.h[i] = Kp[(long)kk * ld + j];
  }
  return f.v;
}

__device__ __forceinline__ v16bf load_b_frag_kv(const unsigned short* __restrict__ V,
                                                int ld, int n0, int s0, int lane) {
  Frag16 f;
  int n  = n0 + (lane & 15);
  int sb = s0 + ((lane & 16) ? 16 : 0);
  if (n < 64) {
#pragma unroll
    for (int i = 0; i < 16; ++i) f.h[i] = V[(long)(sb + i) * ld + n];
  } else if (n == 64) {
#pragma unroll
    for (int i = 0; i < 16; ++i) f.h[i] = 0x3F80;  // bf16 1.0
  } else {
#pragma unroll
    for (int i = 0; i < 16; ++i) f.h[i] = 0;
  }
  return f.v;
}

// ---- async tile staging -----------------------------------------------------
// Copies a 128x32 bf16 tile (rows row0..row0+127, cols k0..k0+31 of a
// row-major matrix) into LDS laid out [128][kLdaS].  256 threads, 16B chunks.
__device__ __forceinline__ void stage_tile(unsigned short* __restrict__ dst,
                                           const unsigned short* __restrict__ src,
                                           long row0, int lda, int k0, int tid) {
#pragma unroll
  for (int i = 0; i < 2; ++i) {
    int c  = tid + 256 * i;        // 0..511: 4 chunks per row, 128 rows
    int r  = c >> 2;
    int ko = (c & 3) * 8;
    const unsigned short* g = src + (row0 + r) * (long)lda + k0 + ko;
    unsigned short* l = dst + r * kLdaS + ko;
#if defined(ASYNC_LDS_OK)
    __builtin_amdgcn_global_load_async_to_lds_b128((gbl_vint4*)g, (lds_vint4*)l, 0, 0);
#else
    *(uint4*)l = *(const uint4*)g;
#endif
  }
}

__device__ __forceinline__ void async_wait_all() {
#if defined(ASYNC_LDS_OK)
  __builtin_amdgcn_s_wait_asynccnt(0);
#endif
}

// ---------------------------------------------------------------------------
// Double-buffered 128x128 block-tile GEMM core (NT): C = A * W^T.
// 8 waves: 4 along M x 2 along N; wave tile 32x64 -> 8 wmma / K-step.
// ---------------------------------------------------------------------------
template <typename Epi>
__device__ __forceinline__ void gemm128x128_staged(
    const unsigned short* __restrict__ A, long arow0, int lda,
    const unsigned short* __restrict__ W, int nrow0, int ldw,
    int K, Epi epi) {
  __shared__ unsigned short As[2][128 * kLdaS];
  __shared__ unsigned short Bs[2][128 * kLdaS];
  const int tid = threadIdx.x;
  const int lane = tid & 31, wid = tid >> 5;
  const int wm = wid & 3, wn = wid >> 2;
  const v8f vz = {0.f, 0.f, 0.f, 0.f, 0.f, 0.f, 0.f, 0.f};
  v8f acc[2][4];
#pragma unroll
  for (int i = 0; i < 2; ++i)
#pragma unroll
    for (int j = 0; j < 4; ++j) acc[i][j] = vz;

  stage_tile(As[0], A, arow0, lda, 0, tid);
  stage_tile(Bs[0], W, nrow0, ldw, 0, tid);

  const int KT = K / 32;
  for (int kt = 0; kt < KT; ++kt) {
    const int buf = kt & 1;
    async_wait_all();        // our own staged copies landed
    __syncthreads();         // everyone's copies visible; prev buffer consumed
    if (kt + 1 < KT) {
      stage_tile(As[buf ^ 1], A, arow0, lda, (kt + 1) * 32, tid);
      stage_tile(Bs[buf ^ 1], W, nrow0, ldw, (kt + 1) * 32, tid);
    }
    v16bf af[2], bf[4];
#pragma unroll
    for (int i = 0; i < 2; ++i) af[i] = lds_a_frag(As[buf], wm * 32 + 16 * i, lane);
#pragma unroll
    for (int j = 0; j < 4; ++j) bf[j] = lds_b_frag(Bs[buf], wn * 64 + 16 * j, lane);
#pragma unroll
    for (int i = 0; i < 2; ++i)
#pragma unroll
      for (int j = 0; j < 4; ++j) acc[i][j] = wmma_bf16(af[i], bf[j], acc[i][j]);
  }
  epi(acc);
}

// ---------------------------------------------------------------------------
// Elementwise prep kernels
// ---------------------------------------------------------------------------
__global__ void cvt_bf16_kernel(const float* __restrict__ in,
                                unsigned short* __restrict__ out, int n) {
  int i = blockIdx.x * 256 + threadIdx.x;
  if (i < n) out[i] = f2bf(in[i]);
}

// x (B,C,P,D) fp32 -> xs (B, S=P*C, D) bf16, s = p*C + c
__global__ void pack_x_kernel(const float* __restrict__ x,
                              unsigned short* __restrict__ xs) {
  long i = (long)blockIdx.x * 256 + threadIdx.x;
  int  d = (int)(i % kD);
  long r = i / kD;
  int  s = (int)(r % kS);
  int  b = (int)(r / kS);
  int  p = s / kC, c = s % kC;
  xs[i] = f2bf(x[(((long)b * kC + c) * kP + p) * kD + d]);
}

// ---------------------------------------------------------------------------
// Cayley: X = inv(I-A)(I+A), A = adj - adj^T (64x64), then
// Peff = projection @ X * sqrt(Dh)/sqrt(m) (bf16).  I-A is positive definite,
// so pivot-free Gauss-Jordan is stable.  One block, 64 threads (col t each).
// ---------------------------------------------------------------------------
__global__ __launch_bounds__(64) void cayley_kernel(
    const float* __restrict__ projection, const float* __restrict__ adj,
    unsigned short* __restrict__ peff) {
  __shared__ float Mm[64][64];
  __shared__ float Xm[64][64];
  const int t = threadIdx.x;
  for (int i = 0; i < 64; ++i) {
    float a = adj[i * 64 + t] - adj[t * 64 + i];
    float d = (i == t) ? 1.0f : 0.0f;
    Mm[i][t] = d - a;
    Xm[i][t] = d + a;
  }
  for (int k = 0; k < 64; ++k) {
    __syncthreads();
    float piv = Mm[k][k];
    float mk = Mm[k][t] / piv;
    float xk = Xm[k][t] / piv;
    __syncthreads();
    Mm[k][t] = mk;
    Xm[k][t] = xk;
    __syncthreads();
    for (int i = 0; i < 64; ++i) {
      float fi = (i == k) ? 0.0f : Mm[i][k];
      __syncthreads();
      Mm[i][t] -= fi * Mm[k][t];
      Xm[i][t] -= fi * Xm[k][t];
    }
  }
  __syncthreads();
  const float scale = sqrtf((float)kDh) * rsqrtf((float)kM);  // == 1 here
  const int r = t;
  for (int c = 0; c < 64; ++c) {
    float acc = 0.f;
    for (int j = 0; j < 64; ++j) acc += projection[r * 64 + j] * Xm[j][c];
    peff[r * 64 + c] = f2bf(acc * scale);
  }
}

// ---------------------------------------------------------------------------
// GEMM1: qkv[row][n] = sum_k xs[row][k] * wqkv[n][k]  (16384 x 3072 x 1024)
// ---------------------------------------------------------------------------
__global__ __launch_bounds__(256) void gemm_qkv_kernel(
    const unsigned short* __restrict__ xs, const unsigned short* __restrict__ wq,
    unsigned short* __restrict__ qkv) {
  const long m0 = (long)blockIdx.x * 128;
  const int  n0 = blockIdx.y * 128;
  const int lane = threadIdx.x & 31, wid = threadIdx.x >> 5;
  const int wm = wid & 3, wn = wid >> 2;
  gemm128x128_staged(xs, m0, kD, wq, n0, kD, kD, [&](v8f (&acc)[2][4]) {
    const int nl = lane & 15, mo = (lane & 16) ? 8 : 0;
#pragma unroll
    for (int i = 0; i < 2; ++i)
#pragma unroll
      for (int j = 0; j < 4; ++j) {
        int col = n0 + wn * 64 + 16 * j + nl;
#pragma unroll
        for (int r = 0; r < 8; ++r) {
          long row = m0 + wm * 32 + 16 * i + mo + r;
          qkv[row * kNqkv + col] = f2bf(acc[i][j][r]);
        }
      }
  });
}

// ---------------------------------------------------------------------------
// phi: out[row][h*64+j] = relu( t[row][h*64+:]@Peff^T ) + eps (K=64, direct).
// ---------------------------------------------------------------------------
__global__ __launch_bounds__(256) void phi_kernel(
    const unsigned short* __restrict__ qkv, int col_off,
    const unsigned short* __restrict__ peff, unsigned short* __restrict__ outp) {
  const int lane = threadIdx.x & 31, wid = threadIdx.x >> 5;
  const long m0 = (long)blockIdx.x * 256 + wid * 32;
  const int  h  = blockIdx.y;
  const unsigned short* A = qkv + col_off + h * kDh;  // lda = kNqkv
  const v8f vz = {0.f, 0.f, 0.f, 0.f, 0.f, 0.f, 0.f, 0.f};
  v8f acc[2][4];
#pragma unroll
  for (int i = 0; i < 2; ++i)
#pragma unroll
    for (int j = 0; j < 4; ++j) acc[i][j] = vz;

  for (int k0 = 0; k0 < kDh; k0 += 32) {
    v16bf a[2], b[4];
#pragma unroll
    for (int i = 0; i < 2; ++i) a[i] = load_a_frag(A, kNqkv, m0 + 16 * i, k0, lane);
#pragma unroll
    for (int j = 0; j < 4; ++j) b[j] = load_b_frag_nt(peff, kDh, 16 * j, k0, lane);
#pragma unroll
    for (int i = 0; i < 2; ++i)
#pragma unroll
      for (int j = 0; j < 4; ++j) acc[i][j] = wmma_bf16(a[i], b[j], acc[i][j]);
  }
  const int nl = lane & 15, mo = (lane & 16) ? 8 : 0;
#pragma unroll
  for (int i = 0; i < 2; ++i)
#pragma unroll
    for (int j = 0; j < 4; ++j) {
      int col = h * kDh + 16 * j + nl;
#pragma unroll
      for (int r = 0; r < 8; ++r) {
        long row = m0 + 16 * i + mo + r;
        outp[row * kD + col] = f2bf(fmaxf(acc[i][j][r], 0.f) + kEps);
      }
    }
}

// ---------------------------------------------------------------------------
// kvs: per (b,h): kvs_ext[j][n] = sum_s kp[s][j] * vext[s][n], n in [0,80),
// ones column at n=64 yields ks_sum.  8 waves reduce via LDS ds_add_f32;
// epilogue stores kvsT bf16 (80 x 64) so the next GEMM is NT.
// ---------------------------------------------------------------------------
__global__ __launch_bounds__(256) void kvs_kernel(
    const unsigned short* __restrict__ kp, const unsigned short* __restrict__ qkv,
    unsigned short* __restrict__ kvsT) {
  __shared__ float lds[64 * 80];
  const int b = blockIdx.x, h = blockIdx.y;
  const int t = threadIdx.x;
  for (int i = t; i < 64 * 80; i += 256) lds[i] = 0.f;
  __syncthreads();

  const int lane = t & 31, wid = t >> 5;
  const int j0 = (wid & 1) * 32;
  const int sc = wid >> 1;
  const unsigned short* Kp = kp  + (long)b * kS * kD + h * kDh;
  const unsigned short* V  = qkv + (long)b * kS * kNqkv + 2 * kD + h * kDh;
  const v8f vz = {0.f, 0.f, 0.f, 0.f, 0.f, 0.f, 0.f, 0.f};
  v8f acc[2][5];
#pragma unroll
  for (int i = 0; i < 2; ++i)
#pragma unroll
    for (int j = 0; j < 5; ++j) acc[i][j] = vz;

  for (int s0 = sc * 1024; s0 < sc * 1024 + 1024; s0 += 32) {
    v16bf a[2], bf[5];
#pragma unroll
    for (int i = 0; i < 2; ++i) a[i] = load_a_frag_t(Kp, kD, j0 + 16 * i, s0, lane);
#pragma unroll
    for (int j = 0; j < 5; ++j) bf[j] = load_b_frag_kv(V, kNqkv, 16 * j, s0, lane);
#pragma unroll
    for (int i = 0; i < 2; ++i)
#pragma unroll
      for (int j = 0; j < 5; ++j) acc[i][j] = wmma_bf16(a[i], bf[j], acc[i][j]);
  }
  const int nl = lane & 15, mo = (lane & 16) ? 8 : 0;
#pragma unroll
  for (int i = 0; i < 2; ++i)
#pragma unroll
    for (int j = 0; j < 5; ++j)
#pragma unroll
      for (int r = 0; r < 8; ++r)
        atomicAdd(&lds[(j0 + 16 * i + mo + r) * 80 + (16 * j + nl)], acc[i][j][r]);
  __syncthreads();

  unsigned short* out = kvsT + (long)(b * kH + h) * 80 * 64;
  for (int idx = t; idx < 80 * 64; idx += 256) {
    int n2 = idx >> 6, j2 = idx & 63;
    out[idx] = f2bf(lds[j2 * 80 + n2]);   // transposed: kvsT[n][j]
  }
}

// ---------------------------------------------------------------------------
// num/den + normalize: y = (qp @ kvs) / (qp @ ks_sum).  N = 80, col 64 = den.
// ---------------------------------------------------------------------------
__global__ __launch_bounds__(256) void attn_out_kernel(
    const unsigned short* __restrict__ qp, const unsigned short* __restrict__ kvsT,
    unsigned short* __restrict__ y) {
  __shared__ float dsh[256];
  const int lane = threadIdx.x & 31, wid = threadIdx.x >> 5;
  const long m0 = (long)blockIdx.x * 256 + wid * 32;
  const int  h  = blockIdx.y;
  const int  b  = (int)(m0 / kS);
  const unsigned short* A = qp + h * kDh;
  const unsigned short* W = kvsT + (long)(b * kH + h) * 80 * 64;
  const v8f vz = {0.f, 0.f, 0.f, 0.f, 0.f, 0.f, 0.f, 0.f};
  v8f acc[2][5];
#pragma unroll
  for (int i = 0; i < 2; ++i)
#pragma unroll
    for (int j = 0; j < 5; ++j) acc[i][j] = vz;

  for (int k0 = 0; k0 < 64; k0 += 32) {
    v16bf a[2], bf[5];
#pragma unroll
    for (int i = 0; i < 2; ++i) a[i] = load_a_frag(A, kD, m0 + 16 * i, k0, lane);
#pragma unroll
    for (int j = 0; j < 5; ++j) bf[j] = load_b_frag_nt(W, 64, 16 * j, k0, lane);
#pragma unroll
    for (int i = 0; i < 2; ++i)
#pragma unroll
      for (int j = 0; j < 5; ++j) acc[i][j] = wmma_bf16(a[i], bf[j], acc[i][j]);
  }
  const int nl = lane & 15, mo = (lane & 16) ? 8 : 0;
  if (nl == 0) {
#pragma unroll
    for (int i = 0; i < 2; ++i)
#pragma unroll
      for (int r = 0; r < 8; ++r)
        dsh[wid * 32 + 16 * i + mo + r] = acc[i][4][r];   // den (col 64)
  }
  __syncthreads();
#pragma unroll
  for (int i = 0; i < 2; ++i)
#pragma unroll
    for (int j = 0; j < 4; ++j) {
      int col = h * kDh + 16 * j + nl;
#pragma unroll
      for (int r = 0; r < 8; ++r) {
        int lrow = wid * 32 + 16 * i + mo + r;
        long row = (long)blockIdx.x * 256 + lrow;
        y[row * kD + col] = f2bf(acc[i][j][r] / dsh[lrow]);
      }
    }
}

// ---------------------------------------------------------------------------
// GEMM3: out = y @ w_out^T + b_out, scattered back to (B,C,P,D) fp32.
// ---------------------------------------------------------------------------
__global__ __launch_bounds__(256) void gemm_out_kernel(
    const unsigned short* __restrict__ y, const unsigned short* __restrict__ wo,
    const float* __restrict__ bias, float* __restrict__ out) {
  const long m0 = (long)blockIdx.x * 128;
  const int  n0 = blockIdx.y * 128;
  const int lane = threadIdx.x & 31, wid = threadIdx.x >> 5;
  const int wm = wid & 3, wn = wid >> 2;
  gemm128x128_staged(y, m0, kD, wo, n0, kD, kD, [&](v8f (&acc)[2][4]) {
    const int nl = lane & 15, mo = (lane & 16) ? 8 : 0;
#pragma unroll
    for (int i = 0; i < 2; ++i)
#pragma unroll
      for (int j = 0; j < 4; ++j) {
        int col = n0 + wn * 64 + 16 * j + nl;
        float bv = bias[col];
#pragma unroll
        for (int r = 0; r < 8; ++r) {
          long row = m0 + wm * 32 + 16 * i + mo + r;
          int bb = (int)(row / kS), s = (int)(row % kS);
          int p = s / kC, c = s % kC;
          out[(((long)bb * kC + c) * kP + p) * kD + col] = acc[i][j][r] + bv;
        }
      }
  });
}

// ---------------------------------------------------------------------------
extern "C" void kernel_launch(void* const* d_in, const int* in_sizes, int n_in,
                              void* d_out, int out_size, void* d_ws, size_t ws_size,
                              hipStream_t stream) {
  (void)in_sizes; (void)n_in; (void)out_size; (void)ws_size;
  const float* x      = (const float*)d_in[0];
  const float* w_qkv  = (const float*)d_in[1];
  const float* w_out  = (const float*)d_in[2];
  const float* b_out  = (const float*)d_in[3];
  const float* proj   = (const float*)d_in[4];
  const float* adj    = (const float*)d_in[5];
  float* out = (float*)d_out;

  char* w = (char*)d_ws;
  size_t off = 0;
  auto take = [&](size_t bytes) {
    size_t o = off; off += (bytes + 255) & ~(size_t)255; return o;
  };
  unsigned short* xs   = (unsigned short*)(w + take((size_t)kRows * kD * 2));    // reused as qp
  unsigned short* qkv  = (unsigned short*)(w + take((size_t)kRows * kNqkv * 2)); // reused as y
  unsigned short* wqb  = (unsigned short*)(w + take((size_t)kNqkv * kD * 2));
  unsigned short* wob  = (unsigned short*)(w + take((size_t)kD * kD * 2));
  unsigned short* peff = (unsigned short*)(w + take((size_t)64 * 64 * 2));
  unsigned short* kp   = (unsigned short*)(w + take((size_t)kRows * kD * 2));
  unsigned short* kvsT = (unsigned short*)(w + take((size_t)kB * kH * 80 * 64 * 2));
  unsigned short* qp = xs;   // xs dead after GEMM1
  unsigned short* y  = qkv;  // q/k/v dead after kvs stage

  cvt_bf16_kernel<<<(kNqkv * kD + 255) / 256, 256, 0, stream>>>(w_qkv, wqb, kNqkv * kD);
  cvt_bf16_kernel<<<(kD * kD + 255) / 256, 256, 0, stream>>>(w_out, wob, kD * kD);
  cayley_kernel<<<1, 64, 0, stream>>>(proj, adj, peff);
  pack_x_kernel<<<(kRows * kD) / 256, 256, 0, stream>>>(x, xs);

  gemm_qkv_kernel<<<dim3(kRows / 128, kNqkv / 128), 256, 0, stream>>>(xs, wqb, qkv);

  phi_kernel<<<dim3(kRows / 256, kH), 256, 0, stream>>>(qkv, 0,  peff, qp);
  phi_kernel<<<dim3(kRows / 256, kH), 256, 0, stream>>>(qkv, kD, peff, kp);

  kvs_kernel<<<dim3(kB, kH), 256, 0, stream>>>(kp, qkv, kvsT);

  attn_out_kernel<<<dim3(kRows / 256, kH), 256, 0, stream>>>(qp, kvsT, y);

  gemm_out_kernel<<<dim3(kRows / 128, kD / 128), 256, 0, stream>>>(y, wob, b_out, out);
}
